// UpStream_53498112639544
// MI455X (gfx1250) — matrix operand order repaired and checked by reading, not compiled, not run on hardware
//
#include <hip/hip_runtime.h>
#include <hip/hip_bf16.h>

#define HID 64
#define NND 2560
#define NE  40960
#define NP  409600
#define BN_EPS 1e-5f

typedef __attribute__((ext_vector_type(16))) _Float16 v16h;
typedef __attribute__((ext_vector_type(8)))  _Float16 v8h;
typedef __attribute__((ext_vector_type(4)))  _Float16 v4h;
typedef __attribute__((ext_vector_type(8)))  float    v8f;

// ---- workspace layout (float offsets) ----
#define OFF_W1A   0        // 64x64  W1 @ L0_w[0:64]
#define OFF_W2B   4096     // 64x64  W2 @ L0_w[64:128]
#define OFF_WEC   8192     // 64x64  We @ L0_w[128:192]
#define OFF_C1    12288    // 64  b1@A + L0_b
#define OFF_C2    12352    // 64  b2@B
#define OFF_CE    12416    // 64  be@C
#define OFF_SUM0  12480    // 64
#define OFF_SQ0   12544    // 64
#define OFF_SUM1  12608    // 64
#define OFF_SQ1   12672    // 64
#define OFF_BN0SC 12736    // 64 scale, then 64 shift
#define OFF_BN1SC 12864    // 64 scale, then 64 shift
#define OFF_BPACK 12992    // _Float16[4096] (L1_w in WMMA-B layout) = 2048 floats
#define OFF_A1    15104    // 2560x64
#define OFF_A2    178944   // 2560x64
#define OFF_H     342784   // 409600x64 (h0, overwritten in-place by h1)

__device__ __forceinline__ void atomAddG(float* p, float v) {
  unsafeAtomicAdd(p, v);   // GLOBAL_ATOMIC_ADD_F32
}

// ---------------------------------------------------------------------------
// kprep: fold weights, pack L1_w for WMMA-B layout, zero stats accumulators.
// ---------------------------------------------------------------------------
__global__ void __launch_bounds__(256)
kprep(float* ws, const float* __restrict__ W1, const float* __restrict__ b1,
      const float* __restrict__ W2, const float* __restrict__ b2,
      const float* __restrict__ We, const float* __restrict__ be,
      const float* __restrict__ L0w, const float* __restrict__ L0b,
      const float* __restrict__ L1w) {
  const int t = threadIdx.x;
  ws[OFF_SUM0 + t] = 0.f;  // zeros SUM0,SQ0,SUM1,SQ1 (256 floats)

  for (int idx = t; idx < 3 * 4096; idx += 256) {
    const int mat = idx >> 12;
    const int rc = idx & 4095;
    const int k = rc >> 6, n = rc & 63;
    const float* w  = (mat == 0) ? W1 : (mat == 1) ? W2 : We;
    const float* L0 = L0w + mat * 64 * 64;
    float s = 0.f;
    for (int m = 0; m < 64; ++m) s += w[k * 64 + m] * L0[m * 64 + n];
    ws[mat * 4096 + rc] = s;
  }
  if (t < 192) {
    const int mat = t >> 6, n = t & 63;
    const float* bv = (mat == 0) ? b1 : (mat == 1) ? b2 : be;
    const float* L0 = L0w + mat * 64 * 64;
    float s = (mat == 0) ? L0b[n] : 0.f;
    for (int m = 0; m < 64; ++m) s += bv[m] * L0[m * 64 + n];
    ws[OFF_C1 + t] = s;
  }
  // B operand pack: op = nt*2+ks ; lane n=lane&15, rows K = ks*32 + (lane>>4)*8 + (i&7) + (i>>3)*16
  _Float16* Bp = (_Float16*)(ws + OFF_BPACK);
  for (int idx = t; idx < 4096; idx += 256) {
    const int i = idx & 15;
    const int slot = idx >> 4;
    const int lane = slot & 31;
    const int op = slot >> 5;
    const int nt = op >> 1, ks = op & 1;
    const int k = ks * 32 + (lane >> 4) * 8 + (i & 7) + (i >> 3) * 16;
    const int n = nt * 16 + (lane & 15);
    Bp[idx] = (_Float16)L1w[k * 64 + n];
  }
}

// ---------------------------------------------------------------------------
// knode: x_emb (9-table gather-sum) then a1 = x_emb@W1A + c1, a2 = x_emb@W2B + c2
// ---------------------------------------------------------------------------
__global__ void __launch_bounds__(256)
knode(float* ws, const float* __restrict__ atab, const int* __restrict__ x) {
  const int t = threadIdx.x;
  const int g = t >> 6, col = t & 63;
  const int node = blockIdx.x * 4 + g;
  __shared__ int xl[4][9];
  __shared__ float xe[4][64];
  if (t < 36) {
    const int gg = t / 9, c = t % 9;
    xl[gg][c] = x[(blockIdx.x * 4 + gg) * 9 + c];
  }
  __syncthreads();
  float s = 0.f;
#pragma unroll
  for (int c = 0; c < 9; ++c) s += atab[(c * 64 + xl[g][c]) * 64 + col];
  xe[g][col] = s;
  __syncthreads();
  const float* W1A = ws + OFF_W1A;
  const float* W2B = ws + OFF_W2B;
  float s1 = ws[OFF_C1 + col], s2 = ws[OFF_C2 + col];
  for (int k = 0; k < 64; ++k) {
    const float xv = xe[g][k];
    s1 += xv * W1A[k * 64 + col];
    s2 += xv * W2B[k * 64 + col];
  }
  ws[OFF_A1 + node * 64 + col] = s1;
  ws[OFF_A2 + node * 64 + col] = s2;
}

// ---------------------------------------------------------------------------
// kinit: h0[p] = a1[idx0[p]] + a2[idx1[p]]   (constants already folded in)
// ---------------------------------------------------------------------------
__global__ void __launch_bounds__(256)
kinit(float* ws, const int* __restrict__ idx0, const int* __restrict__ idx1) {
  const int t = threadIdx.x;
  const long p = (long)blockIdx.x * 16 + (t >> 4);
  const int c4 = (t & 15) * 4;
  const int i0 = idx0[p], i1 = idx1[p];
  const float4 va = *(const float4*)(ws + OFF_A1 + (long)i0 * 64 + c4);
  const float4 vb = *(const float4*)(ws + OFF_A2 + (long)i1 * 64 + c4);
  float4 r;
  r.x = va.x + vb.x; r.y = va.y + vb.y; r.z = va.z + vb.z; r.w = va.w + vb.w;
  *(float4*)(ws + OFF_H + p * 64 + c4) = r;
}

// ---------------------------------------------------------------------------
// kedge: c = (bond_emb)@WeC + ce ; scatter-add into h0 at epos (dup edges sum)
// ---------------------------------------------------------------------------
__global__ void __launch_bounds__(256)
kedge(float* ws, const float* __restrict__ btab,
      const int* __restrict__ eattr, const int* __restrict__ epos) {
  const int t = threadIdx.x;
  __shared__ float sW[4096];
  __shared__ float sce[64];
  __shared__ float sea[4][64];
  for (int i = t; i < 4096; i += 256) sW[i] = ws[OFF_WEC + i];
  if (t < 64) sce[t] = ws[OFF_CE + t];
  __syncthreads();
  const int sub = t >> 6, k = t & 63;
  float* h = ws + OFF_H;
  for (int r = 0; r < 8; ++r) {
    const int e = blockIdx.x * 32 + sub * 8 + r;
    const int a0 = eattr[e * 3 + 0];
    const int a1v = eattr[e * 3 + 1];
    const int a2v = eattr[e * 3 + 2];
    sea[sub][k] = btab[a0 * 64 + k] + btab[(8 + a1v) * 64 + k] +
                  btab[(16 + a2v) * 64 + k];
    __syncthreads();
    float cv = sce[k];
    for (int m = 0; m < 64; ++m) cv += sea[sub][m] * sW[m * 64 + k];
    atomAddG(&h[(long)epos[e] * 64 + k], cv);
    __syncthreads();
  }
}

// ---------------------------------------------------------------------------
// kstat0: per-column sum / sumsq of h0 (register -> LDS -> global atomics)
// ---------------------------------------------------------------------------
__global__ void __launch_bounds__(256)
kstat0(float* ws) {
  const float* h = ws + OFF_H;
  const int t = threadIdx.x;
  const int c4 = (t & 15) * 4, rloc = t >> 4;
  float s[4] = {0, 0, 0, 0}, q[4] = {0, 0, 0, 0};
  for (int it = 0; it < 4; ++it) {
    const long r0 = ((long)blockIdx.x * 4 + it) * 128;
    for (int pass = 0; pass < 8; ++pass) {
      const float4 v = *(const float4*)(h + (r0 + pass * 16 + rloc) * 64 + c4);
      s[0] += v.x; q[0] += v.x * v.x;
      s[1] += v.y; q[1] += v.y * v.y;
      s[2] += v.z; q[2] += v.z * v.z;
      s[3] += v.w; q[3] += v.w * v.w;
    }
  }
  __shared__ float ssum[64], ssq[64];
  if (t < 64) { ssum[t] = 0.f; ssq[t] = 0.f; }
  __syncthreads();
#pragma unroll
  for (int j = 0; j < 4; ++j) {
    atomicAdd(&ssum[c4 + j], s[j]);
    atomicAdd(&ssq[c4 + j], q[j]);
  }
  __syncthreads();
  if (t < 64) {
    atomAddG(ws + OFF_SUM0 + t, ssum[t]);
    atomAddG(ws + OFF_SQ0 + t, ssq[t]);
  }
}

// ---------------------------------------------------------------------------
// kbnfin: scale = gamma*rsqrt(var+eps), shift = beta - mu*scale
// ---------------------------------------------------------------------------
__global__ void kbnfin(float* ws, const float* __restrict__ gamma,
                       const float* __restrict__ beta, int soff, int boff) {
  const int t = threadIdx.x;
  if (t < 64) {
    const float inv = 1.f / (float)NP;
    const float mu = ws[soff + t] * inv;
    const float var = ws[soff + 64 + t] * inv - mu * mu;
    const float rs = rsqrtf(var + BN_EPS);
    const float sc = gamma[t] * rs;
    ws[boff + t] = sc;
    ws[boff + 64 + t] = beta[t] - mu * sc;
  }
}

// ---------------------------------------------------------------------------
// kgemm1: h1 = relu(bn0(h0)) @ L1_w + L1_b   (in place), BN1 stats fused.
// f16 WMMA 16x16x32, f32 accumulate. 8 waves/block, 128-row strips, 4 strips.
// ---------------------------------------------------------------------------
__global__ void __launch_bounds__(256)
kgemm1(float* ws, const float* __restrict__ L1b) {
  float* h = ws + OFF_H;
  const _Float16* Bp = (const _Float16*)(ws + OFF_BPACK);
  const int t = threadIdx.x;
  const int lane = t & 31, wave = t >> 5;
  const int hl = lane >> 4, mr = lane & 15;

  v16h bm[8];
#pragma unroll
  for (int o = 0; o < 8; ++o)
    bm[o] = *(const v16h*)(Bp + (o * 32 + lane) * 16);

  float bias[4];
#pragma unroll
  for (int nt = 0; nt < 4; ++nt) bias[nt] = L1b[nt * 16 + mr];

  const int c4 = (t & 15) * 4;
  const int rloc = t >> 4;
  const float4 scl = *(const float4*)(ws + OFF_BN0SC + c4);
  const float4 shf = *(const float4*)(ws + OFF_BN0SC + 64 + c4);

  __shared__ __align__(32) _Float16 tile[128 * 72];
  __shared__ float rsum[64], rsq[64];

  float psum[4] = {0, 0, 0, 0}, psq[4] = {0, 0, 0, 0};

  for (int it = 0; it < 4; ++it) {
    const long r0 = ((long)blockIdx.x * 4 + it) * 128;
    __syncthreads();  // tile WAR vs previous iteration readers
#pragma unroll
    for (int pass = 0; pass < 8; ++pass) {
      const int row = pass * 16 + rloc;
      const float4 v = *(const float4*)(h + (r0 + row) * 64 + c4);
      v4h pk;
      pk[0] = (_Float16)fmaxf(v.x * scl.x + shf.x, 0.f);
      pk[1] = (_Float16)fmaxf(v.y * scl.y + shf.y, 0.f);
      pk[2] = (_Float16)fmaxf(v.z * scl.z + shf.z, 0.f);
      pk[3] = (_Float16)fmaxf(v.w * scl.w + shf.w, 0.f);
      *(v4h*)(tile + row * 72 + c4) = pk;
    }
    __syncthreads();

    const int m0 = wave * 16;
    v8f acc[4];
#pragma unroll
    for (int nt = 0; nt < 4; ++nt)
#pragma unroll
      for (int i = 0; i < 8; ++i) acc[nt][i] = bias[nt];

#pragma unroll
    for (int ks = 0; ks < 2; ++ks) {
      const _Float16* ap = tile + (m0 + mr) * 72 + ks * 32 + hl * 8;
      const v8h lo = *(const v8h*)ap;
      const v8h hi = *(const v8h*)(ap + 16);
      v16h a;
#pragma unroll
      for (int i = 0; i < 8; ++i) { a[i] = lo[i]; a[i + 8] = hi[i]; }
#pragma unroll
      for (int nt = 0; nt < 4; ++nt)
        acc[nt] = __builtin_amdgcn_wmma_f32_16x16x32_f16(
            false, a, false, bm[nt * 2 + ks], (short)0, acc[nt], false, false);
    }

#pragma unroll
    for (int nt = 0; nt < 4; ++nt) {
      const int col = nt * 16 + mr;
#pragma unroll
      for (int v = 0; v < 8; ++v) {
        const float val = acc[nt][v];
        psum[nt] += val;
        psq[nt] += val * val;
        h[(r0 + m0 + v + hl * 8) * 64 + col] = val;  // in-place h1
      }
    }
  }

  if (t < 64) { rsum[t] = 0.f; rsq[t] = 0.f; }
  __syncthreads();
#pragma unroll
  for (int nt = 0; nt < 4; ++nt) {
    atomicAdd(&rsum[nt * 16 + mr], psum[nt]);
    atomicAdd(&rsq[nt * 16 + mr], psq[nt]);
  }
  __syncthreads();
  if (t < 64) {
    atomAddG(ws + OFF_SUM1 + t, rsum[t]);
    atomAddG(ws + OFF_SQ1 + t, rsq[t]);
  }
}

// ---------------------------------------------------------------------------
// kout: out[p] = relu(bn1(h1[p])) . L2_w + L2_b
// ---------------------------------------------------------------------------
__global__ void __launch_bounds__(256)
kout(const float* __restrict__ ws, const float* __restrict__ L2w,
     const float* __restrict__ L2b, float* __restrict__ out) {
  const float* h = ws + OFF_H;
  const float* sc = ws + OFF_BN1SC;
  const float* sh = ws + OFF_BN1SC + 64;
  const int t = threadIdx.x;
  const int row = t >> 2, seg = t & 3;
  const long p = (long)blockIdx.x * 64 + row;
  const float* hp = h + p * 64 + seg * 16;
  float s = 0.f;
#pragma unroll
  for (int j = 0; j < 16; ++j) {
    const int c = seg * 16 + j;
    const float v = fmaxf(hp[j] * sc[c] + sh[c], 0.f);
    s += v * L2w[c];
  }
  __shared__ float part[64][4];
  part[row][seg] = s;
  __syncthreads();
  if (t < 64)
    out[(long)blockIdx.x * 64 + t] =
        part[t][0] + part[t][1] + part[t][2] + part[t][3] + L2b[0];
}

// ---------------------------------------------------------------------------
extern "C" void kernel_launch(void* const* d_in, const int* in_sizes, int n_in,
                              void* d_out, int out_size, void* d_ws,
                              size_t ws_size, hipStream_t stream) {
  (void)in_sizes; (void)n_in; (void)out_size; (void)ws_size;
  const float* atab  = (const float*)d_in[0];
  const float* btab  = (const float*)d_in[1];
  const float* W1    = (const float*)d_in[2];
  const float* b1    = (const float*)d_in[3];
  const float* W2    = (const float*)d_in[4];
  const float* b2    = (const float*)d_in[5];
  const float* We    = (const float*)d_in[6];
  const float* be    = (const float*)d_in[7];
  const float* L0w   = (const float*)d_in[8];
  const float* L0b   = (const float*)d_in[9];
  const float* L1w   = (const float*)d_in[10];
  const float* L1b   = (const float*)d_in[11];
  const float* L2w   = (const float*)d_in[12];
  const float* L2b   = (const float*)d_in[13];
  const float* g0    = (const float*)d_in[14];
  const float* beta0 = (const float*)d_in[15];
  const float* g1    = (const float*)d_in[16];
  const float* beta1 = (const float*)d_in[17];
  const int* x     = (const int*)d_in[18];
  const int* eattr = (const int*)d_in[19];
  const int* idx0  = (const int*)d_in[20];
  const int* idx1  = (const int*)d_in[21];
  const int* epos  = (const int*)d_in[22];
  float* ws = (float*)d_ws;
  float* out = (float*)d_out;

  kprep<<<1, 256, 0, stream>>>(ws, W1, b1, W2, b2, We, be, L0w, L0b, L1w);
  knode<<<NND / 4, 256, 0, stream>>>(ws, atab, x);
  kinit<<<NP / 16, 256, 0, stream>>>(ws, idx0, idx1);
  kedge<<<NE / 32, 256, 0, stream>>>(ws, btab, eattr, epos);
  kstat0<<<800, 256, 0, stream>>>(ws);
  kbnfin<<<1, 64, 0, stream>>>(ws, g0, beta0, OFF_SUM0, OFF_BN0SC);
  kgemm1<<<800, 256, 0, stream>>>(ws, L1b);
  kbnfin<<<1, 64, 0, stream>>>(ws, g1, beta1, OFF_SUM1, OFF_BN1SC);
  kout<<<NP / 64, 256, 0, stream>>>(ws, L2w, L2b, out);
}
